// TeixidoBorealUnit_56341380989022
// MI455X (gfx1250) — compile-verified, matrix operand order checked by reading, not run on hardware
//
#include <hip/hip_runtime.h>

#define BATCH 128
#define N_IN  2048
#define N_OUT 1024
// EPSILON = 1.0, TEMPERATURE = 0.1, NEG_FILL = -1e9

typedef __attribute__((ext_vector_type(16))) _Float16 v16h;
typedef __attribute__((ext_vector_type(8)))  _Float16 v8h;
typedef __attribute__((ext_vector_type(8)))  float    v8f;

// monotonic float -> uint key (preserves total order)
__device__ __forceinline__ unsigned f2key(float f) {
    unsigned u = __float_as_uint(f);
    return (u & 0x80000000u) ? ~u : (u | 0x80000000u);
}

// Kernel 1: one block per batch row.
//  - exact lower median (rank (N_IN-1)/2 = 1023) via 4-pass radix select
//  - gate, per-row max of gated value
//  - A'[b,i] = exp((gated - rowmax) * 10) stored as f16
__global__ __launch_bounds__(256) void median_gate_exp_kernel(
    const float* __restrict__ x, _Float16* __restrict__ A, float* __restrict__ rowmax)
{
    const int b   = blockIdx.x;
    const int tid = threadIdx.x;
    const float* xrow = x + (size_t)b * N_IN;

    __shared__ unsigned hist[256];
    __shared__ float    sred[256];
    __shared__ unsigned s_prefix, s_rank;

    if (tid == 0) { s_prefix = 0u; s_rank = 1023u; }   // rank = (N_IN-1)//2
    __syncthreads();

    // 4 x 8-bit MSB-first radix select for the exact rank-1023 key
    for (int pass = 0; pass < 4; ++pass) {
        hist[tid] = 0u;
        __syncthreads();
        const int      shift  = 24 - 8 * pass;
        const unsigned prefix = s_prefix;
        const unsigned himask = (pass == 0) ? 0u : (0xFFFFFFFFu << (32 - 8 * pass));
        for (int j = 0; j < N_IN / 256; ++j) {
            unsigned k = f2key(xrow[tid + j * 256]);
            if ((k & himask) == prefix)
                atomicAdd(&hist[(k >> shift) & 0xFFu], 1u);
        }
        __syncthreads();
        if (tid == 0) {
            unsigned rank = s_rank, cum = 0u, d = 0u;
            for (; d < 255u; ++d) {
                unsigned c = hist[d];
                if (cum + c > rank) break;
                cum += c;
            }
            s_prefix = prefix | (d << shift);
            s_rank   = rank - cum;
        }
        __syncthreads();
    }

    const unsigned km = s_prefix;
    const float cons = __uint_as_float((km & 0x80000000u) ? (km & 0x7FFFFFFFu) : ~km);

    float g[N_IN / 256];
    float m = -3.402823466e38f;
    for (int j = 0; j < N_IN / 256; ++j) {
        float v  = xrow[tid + j * 256];
        float gg = (fabsf(v - cons) < 1.0f) ? v : 0.0f;   // gated_x = x * gate
        g[j] = gg;
        m = fmaxf(m, gg);
    }
    sred[tid] = m;
    __syncthreads();
    for (int s = 128; s > 0; s >>= 1) {
        if (tid < s) sred[tid] = fmaxf(sred[tid], sred[tid + s]);
        __syncthreads();
    }
    const float M = sred[0];
    if (tid == 0) rowmax[b] = M;

    const float C = 14.426950408889634f;  // (1/T)/ln2 = 10*log2(e)
    for (int j = 0; j < N_IN / 256; ++j)
        A[(size_t)b * N_IN + tid + j * 256] = (_Float16)exp2f((g[j] - M) * C);
}

// Kernel 2: B[o,i] = mask ? exp(w*10) : 0, f16, row-major over K (= input layout)
__global__ __launch_bounds__(256) void weights_exp_kernel(
    const float* __restrict__ w, const float* __restrict__ mask, _Float16* __restrict__ Bt)
{
    const int i = blockIdx.x * 256 + threadIdx.x;   // exact multiple, no guard
    float mk = mask[i];
    float wv = w[i];
    Bt[i] = (mk == 0.0f) ? (_Float16)0.0f
                         : (_Float16)exp2f(wv * 14.426950408889634f);
}

// Kernel 3: 128x1024 = A'[128x2048] * B^T[2048x1024] via v_wmma_f32_16x16x32_f16,
// then out = rowmax + T*log(S). One wave per 16x16 tile, 8 waves/block, grid 8x8.
__global__ __launch_bounds__(256) void gemm_lse_kernel(
    const _Float16* __restrict__ A, const _Float16* __restrict__ Bt,
    const float* __restrict__ rowmax, float* __restrict__ out)
{
    const int wave  = threadIdx.x >> 5;
    const int lane  = threadIdx.x & 31;
    const int nTile = blockIdx.x * 8 + wave;   // 0..63
    const int mTile = blockIdx.y;              // 0..7
    const int mBase = mTile * 16, nBase = nTile * 16;
    const int r16 = lane & 15, hi = lane >> 4;

    // ISA 16-bit A layout: lane holds row (lane&15); K chunks [hi*8, hi*8+8)
    // in v0..3 and [hi*8+16, hi*8+24) in v4..7 -> two contiguous b128 loads.
    const _Float16* aPtr = A  + (size_t)(mBase + r16) * N_IN + hi * 8;
    const _Float16* bPtr = Bt + (size_t)(nBase + r16) * N_IN + hi * 8;

    v8f c = {};
#pragma unroll 4
    for (int k = 0; k < N_IN; k += 32) {
        v8h a0 = *(const v8h*)(aPtr + k);
        v8h a1 = *(const v8h*)(aPtr + k + 16);
        v8h b0 = *(const v8h*)(bPtr + k);
        v8h b1 = *(const v8h*)(bPtr + k + 16);
        v16h av = __builtin_shufflevector(a0, a1, 0,1,2,3,4,5,6,7,8,9,10,11,12,13,14,15);
        v16h bv = __builtin_shufflevector(b0, b1, 0,1,2,3,4,5,6,7,8,9,10,11,12,13,14,15);
        c = __builtin_amdgcn_wmma_f32_16x16x32_f16(
                /*neg_a=*/false, av, /*neg_b=*/false, bv,
                /*c_mod=*/(short)0, c, /*reuse_a=*/false, /*reuse_b=*/false);
    }

    // C/D layout: lane<16 -> v_r = (M=r, N=lane); lane>=16 -> v_r = (M=8+r, N=lane-16)
    const int col = nBase + r16;
#pragma unroll
    for (int r = 0; r < 8; ++r) {
        const int row = mBase + hi * 8 + r;
        float s = c[r];
        float o = (s > 0.0f)
                    ? (rowmax[row] + 0.1f * logf(s))
                    : (-1.0e9f + 0.1f * 7.62461898f);   // all-masked row fallback
        out[(size_t)row * N_OUT + col] = o;
    }
}

extern "C" void kernel_launch(void* const* d_in, const int* in_sizes, int n_in,
                              void* d_out, int out_size, void* d_ws, size_t ws_size,
                              hipStream_t stream) {
    const float* x    = (const float*)d_in[0];   // [128, 2048]
    const float* w    = (const float*)d_in[1];   // [1024, 2048]
    const float* mask = (const float*)d_in[2];   // [1024, 2048]
    float* out = (float*)d_out;                  // [128, 1024] f32

    char* ws = (char*)d_ws;
    float*    rowmax = (float*)ws;                                            // 512 B
    _Float16* A      = (_Float16*)(ws + 512);                                 // 512 KB
    _Float16* Bt     = (_Float16*)(ws + 512 + (size_t)BATCH * N_IN * 2);      // 4 MB

    median_gate_exp_kernel<<<BATCH, 256, 0, stream>>>(x, A, rowmax);
    weights_exp_kernel<<<(N_OUT * N_IN) / 256, 256, 0, stream>>>(w, mask, Bt);
    gemm_lse_kernel<<<dim3(8, 8), 256, 0, stream>>>(A, Bt, rowmax, out);

    (void)in_sizes; (void)n_in; (void)out_size; (void)ws_size;
}
